// AttentiveQuantizer_57088705299029
// MI455X (gfx1250) — matrix operand order repaired and checked by reading, not compiled, not run on hardware
//
#include <hip/hip_runtime.h>

// ---------------------------------------------------------------------------
// AttentiveQuantizer for MI455X (gfx1250), wave32, WMMA f16 16x16x32.
// S=4 splits, K=4096 codes, D=128, N=4, H=W=32, hw=1024, M=hw*N=4096.
//
// Kernel A: Q/K/V projection GEMMs (WMMA, f32->f16).
// Kernel B: fused logits GEMM + gumbel-softmax + hard gather + soft@V GEMM.
//           Logits stripe lives in LDS (128 KB of the WGP's 320 KB), is
//           written to HBM exactly once (non-temporal) and never re-read.
// ---------------------------------------------------------------------------

typedef __attribute__((ext_vector_type(16))) _Float16 f16x16;
typedef __attribute__((ext_vector_type(8)))  float    f32x8;

#define SPLITS 4
#define KCODES 4096
#define DDIM   128
#define NBATCH 4
#define HWDIM  1024
#define MROWS  4096            // hw * N
#define CCH    512             // SPLITS * DDIM

// d_out layout (flat f32): quantized | softQ | logits
#define QOFF   0
#define SQOFF  (2097152)
#define LOFF   (4194304)

// ---------------- WMMA helpers (contiguous k order: k = 16*hi + s, A and B alike) --------------

__device__ inline f32x8 wmma_f16(f16x16 a, f16x16 b, f32x8 c) {
  return __builtin_amdgcn_wmma_f32_16x16x32_f16(
      /*neg_a=*/false, a, /*neg_b=*/false, b,
      /*c_mod=*/(short)0, c, /*reuse_a=*/false, /*reuse_b=*/false);
}

// A-fragment: 16x32 (MxK), source row-major [row][k], lda elements.
// lane<16 : row=lane,    K = 0..15
// lane>=16: row=lane-16, K = 16..31
__device__ inline f16x16 load_a_frag(const _Float16* base, int lda,
                                     int row0, int kc, int lane) {
  int l = lane & 15, hi = lane >> 4;
  const _Float16* p0 = base + (size_t)(row0 + l) * lda + kc + hi * 16;
  const _Float16* p1 = p0 + 8;
  f16x16 a;
#pragma unroll
  for (int e = 0; e < 8; ++e) { a[e] = p0[e]; a[e + 8] = p1[e]; }
  return a;
}

// B-fragment: 32x16 (KxN), source stored "by column": baseT[n][k], ldb elems.
// lane<16: N=lane, K=0..15 ; lane>=16: N=lane-16, K=16..31
__device__ inline f16x16 load_b_frag(const _Float16* baseT, int ldb,
                                     int n0, int kc, int lane) {
  int l = lane & 15, hi = lane >> 4;
  const _Float16* p = baseT + (size_t)(n0 + l) * ldb + kc + hi * 16;
  f16x16 b;
#pragma unroll
  for (int e = 0; e < 16; ++e) b[e] = p[e];
  return b;
}

// Same as load_b_frag but converts from f32 source (weights).
__device__ inline f16x16 load_b_frag_f32(const float* baseT, int ldb,
                                         int n0, int kc, int lane) {
  int l = lane & 15, hi = lane >> 4;
  const float* p = baseT + (size_t)(n0 + l) * ldb + kc + hi * 16;
  f16x16 b;
#pragma unroll
  for (int e = 0; e < 16; ++e) b[e] = (_Float16)p[e];
  return b;
}

// ---------------------------------------------------------------------------
// Kernel A: projections.  grid.x = 3 * S * (4096/16) = 3072, block = 256.
//   g==0: query  = xs @ Wq^T + bq   (rows from latent, permuted gather)
//   g==1: keyp   = cb @ Wk^T + bk
//   g==2: value  = cb @ Wv^T + bv   (stored f32 + transposed f16)
// ---------------------------------------------------------------------------
__global__ __launch_bounds__(256)
void aq_proj_kernel(const float* __restrict__ latent,
                    const float* __restrict__ codebooks,
                    const float* __restrict__ Wq, const float* __restrict__ bq,
                    const float* __restrict__ Wk, const float* __restrict__ bk,
                    const float* __restrict__ Wv, const float* __restrict__ bv,
                    _Float16* __restrict__ Qh, _Float16* __restrict__ Kh,
                    _Float16* __restrict__ VhT, float* __restrict__ Vf) {
  __shared__ _Float16 Atile[16 * DDIM];

  const int wg  = blockIdx.x;
  const int mt  = wg & 255;          // 16-row tile index
  const int s   = (wg >> 8) & 3;
  const int g   = wg >> 10;          // 0=q 1=k 2=v
  const int tid = threadIdx.x;
  const int lane = tid & 31, wave = tid >> 5;
  const int m0 = mt * 16;

  // Stage A-tile (16 x 128) into LDS as f16.
  for (int i = tid; i < 16 * DDIM; i += 256) {
    int r = i >> 7, d = i & 127;
    float v;
    if (g == 0) {
      int m = m0 + r, p = m >> 2, n = m & 3;          // m = p*N + n
      v = __builtin_nontemporal_load(
          &latent[(size_t)(n * CCH + s * DDIM + d) * HWDIM + p]);
    } else {
      v = codebooks[((size_t)s * KCODES + m0 + r) * DDIM + d];
    }
    Atile[i] = (_Float16)v;
  }
  __syncthreads();

  const float* W    = (g == 0 ? Wq : (g == 1 ? Wk : Wv)) + (size_t)s * DDIM * DDIM;
  const float* bias = (g == 0 ? bq : (g == 1 ? bk : bv)) + s * DDIM;

  const int e0 = wave * 16;          // 8 waves cover D=128
  f32x8 c = {};
#pragma unroll
  for (int kc = 0; kc < DDIM; kc += 32) {
    f16x16 a = load_a_frag(Atile, DDIM, 0, kc, lane);
    f16x16 b = load_b_frag_f32(W, DDIM, e0, kc, lane);   // B[n=e_out][k=d]
    c = wmma_f16(a, b, c);
  }

  const int l = lane & 15, hi = lane >> 4;
  const float badd = bias[e0 + l];
#pragma unroll
  for (int r = 0; r < 8; ++r) {
    int row = m0 + r + hi * 8;
    int col = e0 + l;
    float val = c[r] + badd;
    if (g == 0) {
      Qh[((size_t)s * MROWS + row) * DDIM + col] = (_Float16)val;
    } else if (g == 1) {
      Kh[((size_t)s * KCODES + row) * DDIM + col] = (_Float16)val;
    } else {
      Vf[((size_t)s * KCODES + row) * DDIM + col] = val;
      VhT[((size_t)s * DDIM + col) * KCODES + row] = (_Float16)val;
    }
  }
}

// ---------------------------------------------------------------------------
// Kernel B: fused  logits -> gumbel-softmax -> {argmax gather, soft@V}.
// grid.x = S * 256 (one WG per 16-row M-stripe), block = 256 (8 waves).
//
// Phase 1: 8 waves x 32 N-tiles: logits = Q-tile @ Kh^T / 64.
//          f32 result -> d_out (non-temporal); f16 copy -> LDS stripe.
// Phase 2: t = (logit + gumbel)/temp in LDS; rowwise max/argmax; exp; sum.
// Phase 3: quantized = Vf[argmax]; softQ = (exp-stripe @ VhT) / rowsum.
// ---------------------------------------------------------------------------
__global__ __launch_bounds__(256)
void aq_fused_kernel(const _Float16* __restrict__ Qh,
                     const _Float16* __restrict__ Kh,
                     const float* __restrict__ gumbel_u,
                     const float* __restrict__ temp_p,
                     const _Float16* __restrict__ VhT,
                     const float* __restrict__ Vf,
                     float* __restrict__ out) {
  __shared__ _Float16 tbuf[16 * KCODES];     // logits -> t -> exp(t-max), f16
  __shared__ _Float16 Qt[16 * DDIM];
  __shared__ float  rpart[16 * 16];
  __shared__ int    ipart[16 * 16];
  __shared__ float  rowsum[16];
  __shared__ int    rowidx[16];

  const int wg = blockIdx.x;
  const int mt = wg & 255;
  const int s  = wg >> 8;
  const int tid = threadIdx.x;
  const int lane = tid & 31, wave = tid >> 5;
  const int l = lane & 15, hi = lane >> 4;
  const int m0 = mt * 16;
  const float inv_temp = 1.0f / temp_p[0];

  // Stage Q-tile (16 x 128).
  for (int i = tid; i < 16 * DDIM; i += 256)
    Qt[i] = Qh[((size_t)s * MROWS + m0 + (i >> 7)) * DDIM + (i & 127)];
  __syncthreads();

  // ---- Phase 1: logits GEMM. Q A-fragments are loop-invariant: hoist. ----
  f16x16 afrag[4];
#pragma unroll
  for (int j = 0; j < 4; ++j) afrag[j] = load_a_frag(Qt, DDIM, 0, j * 32, lane);

  const _Float16* Kbase = Kh + (size_t)s * KCODES * DDIM;
  for (int it = 0; it < 32; ++it) {
    int k0 = (wave * 32 + it) * 16;          // this wave's codeword tile
    f32x8 acc = {};
#pragma unroll
    for (int j = 0; j < 4; ++j) {
      f16x16 b = load_b_frag(Kbase, DDIM, k0, j * 32, lane); // B[n=code][k=e]
      acc = wmma_f16(afrag[j], b, acc);
    }
#pragma unroll
    for (int r = 0; r < 8; ++r) {
      int row = r + hi * 8;
      int m = m0 + row, p = m >> 2, n = m & 3;
      int code = k0 + l;
      float lg = acc[r] * 0.015625f;         // 1/sqrt(4096)
      __builtin_nontemporal_store(
          lg, &out[LOFF + ((size_t)((s * NBATCH + n) * HWDIM + p)) * KCODES + code]);
      tbuf[row * KCODES + code] = (_Float16)lg;
    }
  }
  __syncthreads();

  // ---- Phase 2: gumbel + softmax over the LDS stripe. 16 threads/row. ----
  const int r = tid >> 4, sub = tid & 15;
  {
    const int m = m0 + r, p = m >> 2, n = m & 3;
    const float* Grow = gumbel_u +
        ((size_t)((s * HWDIM + p) * NBATCH + n)) * KCODES;

    float bmax = -__builtin_inff();
    int   bidx = 0;
    for (int k = sub; k < KCODES; k += 16) {
      float u = __builtin_nontemporal_load(Grow + k);
      float g = -__logf(-__logf(u));
      float t = ((float)tbuf[r * KCODES + k] + g) * inv_temp;
      tbuf[r * KCODES + k] = (_Float16)t;
      if (t > bmax) { bmax = t; bidx = k; }  // strictly >, keeps first index
    }
    rpart[r * 16 + sub] = bmax;
    ipart[r * 16 + sub] = bidx;
  }
  __syncthreads();

  float rmax = -__builtin_inff();
  int   ridx = 0x7fffffff;
#pragma unroll
  for (int j = 0; j < 16; ++j) {
    float v = rpart[r * 16 + j];
    int   id = ipart[r * 16 + j];
    if (v > rmax || (v == rmax && id < ridx)) { rmax = v; ridx = id; }
  }
  __syncthreads();

  float sum = 0.0f;
  for (int k = sub; k < KCODES; k += 16) {
    float t = (float)tbuf[r * KCODES + k];
    float e = __expf(t - rmax);
    tbuf[r * KCODES + k] = (_Float16)e;
    sum += e;
  }
  rpart[r * 16 + sub] = sum;
  if (sub == 0) rowidx[r] = ridx;
  __syncthreads();
  if (sub == 0) {
    float ss = 0.0f;
#pragma unroll
    for (int j = 0; j < 16; ++j) ss += rpart[r * 16 + j];
    rowsum[r] = ss;
  }
  __syncthreads();

  // ---- Phase 3a: quantized = V[argmax] (hard ST forward), f32 gather. ----
  for (int i = tid; i < 16 * DDIM; i += 256) {
    int rr = i >> 7, e = i & 127;
    int idx = rowidx[rr];
    float val = Vf[((size_t)s * KCODES + idx) * DDIM + e];
    int mm = m0 + rr, pp = mm >> 2, nn = mm & 3;
    __builtin_nontemporal_store(
        val, &out[QOFF + (size_t)(nn * CCH + s * DDIM + e) * HWDIM + pp]);
  }

  // ---- Phase 3b: softQ = exp-stripe @ VhT, normalized per row. ----------
  const int e0 = wave * 16;                  // 8 waves cover D=128
  const _Float16* Vt = VhT + (size_t)s * DDIM * KCODES;  // [e][k]
  f32x8 acc = {};
  for (int kc = 0; kc < KCODES; kc += 32) {
    f16x16 a = load_a_frag(tbuf, KCODES, 0, kc, lane);
    f16x16 b = load_b_frag(Vt, KCODES, e0, kc, lane);
    acc = wmma_f16(a, b, acc);
  }
#pragma unroll
  for (int rr = 0; rr < 8; ++rr) {
    int rl = rr + hi * 8;
    float v = acc[rr] / rowsum[rl];
    int mm = m0 + rl, pp = mm >> 2, nn = mm & 3;
    int e = e0 + l;
    __builtin_nontemporal_store(
        v, &out[SQOFF + (size_t)(nn * CCH + s * DDIM + e) * HWDIM + pp]);
  }
}

// ---------------------------------------------------------------------------
extern "C" void kernel_launch(void* const* d_in, const int* in_sizes, int n_in,
                              void* d_out, int out_size, void* d_ws,
                              size_t ws_size, hipStream_t stream) {
  const float* latent    = (const float*)d_in[0];
  const float* codebooks = (const float*)d_in[1];
  const float* Wq = (const float*)d_in[2];
  const float* bq = (const float*)d_in[3];
  const float* Wk = (const float*)d_in[4];
  const float* bk = (const float*)d_in[5];
  const float* Wv = (const float*)d_in[6];
  const float* bv = (const float*)d_in[7];
  const float* gumbel_u = (const float*)d_in[8];
  const float* temp_p   = (const float*)d_in[9];
  float* out = (float*)d_out;

  // Workspace: Qh(4MiB f16) | Kh(4MiB f16) | VhT(4MiB f16) | Vf(8MiB f32)
  char* ws = (char*)d_ws;
  _Float16* Qh  = (_Float16*)(ws + 0);
  _Float16* Kh  = (_Float16*)(ws + (size_t)4 * 1024 * 1024);
  _Float16* VhT = (_Float16*)(ws + (size_t)8 * 1024 * 1024);
  float*    Vf  = (float*)   (ws + (size_t)12 * 1024 * 1024);

  aq_proj_kernel<<<3 * SPLITS * (MROWS / 16), 256, 0, stream>>>(
      latent, codebooks, Wq, bq, Wk, bk, Wv, bv, Qh, Kh, VhT, Vf);

  aq_fused_kernel<<<SPLITS * (MROWS / 16), 256, 0, stream>>>(
      Qh, Kh, gumbel_u, temp_p, VhT, Vf, out);
}
